// ProbAttention_79826262163770
// MI455X (gfx1250) — matrix operand order repaired and checked by reading, not compiled, not run on hardware
//
#include <hip/hip_runtime.h>

typedef __attribute__((ext_vector_type(16))) _Float16 v16h;
typedef __attribute__((ext_vector_type(8)))  _Float16 v8h;
typedef __attribute__((ext_vector_type(8)))  float    v8f;
typedef __attribute__((ext_vector_type(4)))  float    f4;

#define L_SEQ   2048
#define D_MODEL 512
#define NHEAD   8
#define DK      64
#define N_U     40
#define SAMPK   40
#define NBH     16            // B * H
#define NROWS   (2 * L_SEQ)   // B * L = 4096
#define CTX_ELEMS (2 * NHEAD * L_SEQ * DK)          // 2,097,152
#define ATTNS_ELEMS ((size_t)NBH * L_SEQ * L_SEQ)   // 67,108,864

// ---------- fragment helpers (CDNA5 wave32 WMMA layouts) ----------
// A 16x32 f16: lane holds row m=lane%16; halves 0..15 = K(k0+16*hi ..).
// B 32x16 f16: lane holds column
// n=lane%16; halves 0..15 = K(k0+16*hi ..).
__device__ __forceinline__ v16h cat8(v8h lo, v8h hi) {
  return __builtin_shufflevector(lo, hi, 0, 1, 2, 3, 4, 5, 6, 7,
                                 8, 9, 10, 11, 12, 13, 14, 15);
}
__device__ __forceinline__ v16h load_h16(const _Float16* __restrict__ p) {
  v8h lo = *(const v8h*)p;
  v8h hi = *(const v8h*)(p + 8);
  return cat8(lo, hi);
}

// CDNA5 async global->LDS copy (16B per lane), tracked by ASYNCcnt.
__device__ __forceinline__ void async_load_b128(unsigned lds_addr,
                                                const void* gptr) {
  asm volatile("global_load_async_to_lds_b128 %0, %1, off"
               :: "v"(lds_addr), "v"(gptr) : "memory");
}
__device__ __forceinline__ void wait_asynccnt0() {
  asm volatile("s_wait_asynccnt 0x0" ::: "memory");
}

// ---- 0) one-shot f32 -> f16 conversion ----
__global__ void __launch_bounds__(256)
cvt_f16_kernel(const float* __restrict__ src, _Float16* __restrict__ dst, int n) {
  const int i = blockIdx.x * 256 + threadIdx.x;
  if (i < n) dst[i] = (_Float16)src[i];
}

// ---- 1) Projection GEMM: out(B,H,L,dk) = x(B*L,D) @ W(D,D)^T + bias ----
// 128 threads = 4 waves; A tile (16 x 512 f16, 16KB) staged once to LDS via
// async copies and shared by all 4 waves; each wave owns a 16(M) x 64(N)
// strip. B fragments are software-pipelined: iteration k prefetches the
// k+32 fragments before consuming the current ones in 4 WMMAs, so the
// pre-WMMA wait tolerates 8 outstanding loads instead of 0.
__global__ void __launch_bounds__(128)
proj_wmma_kernel(const _Float16* __restrict__ xh, const _Float16* __restrict__ Wh,
                 const float* __restrict__ bias, float* __restrict__ outf,
                 _Float16* __restrict__ outh) {
  __shared__ _Float16 atile[16 * D_MODEL];   // 16 KB
  const int tid  = threadIdx.x;
  const int lane = tid & 31;
  const int wid  = tid >> 5;
  const int hi   = lane >> 4;
  const int ln16 = lane & 15;
  const int m0   = blockIdx.y * 16;                 // row block in [0, B*L)
  const int n0   = blockIdx.x * 256 + wid * 64;     // per-wave column strip

  // stage A tile: 16 rows x 1024B = 1024 chunks of 16B, 128 threads x 8 iters
  {
    const char* src = (const char*)(xh + (size_t)m0 * D_MODEL);
    const unsigned lds_base = (unsigned)(size_t)(&atile[0]);
#pragma unroll
    for (int i = 0; i < 8; ++i) {
      const int c = i * 128 + tid;
      async_load_b128(lds_base + (unsigned)(c * 16), src + (size_t)c * 16);
    }
    wait_asynccnt0();
    __syncthreads();
  }

  const _Float16* arow = &atile[ln16 * D_MODEL];
  const _Float16* brow0 = Wh + (size_t)(n0 +  0 + ln16) * D_MODEL + 16 * hi;
  const _Float16* brow1 = Wh + (size_t)(n0 + 16 + ln16) * D_MODEL + 16 * hi;
  const _Float16* brow2 = Wh + (size_t)(n0 + 32 + ln16) * D_MODEL + 16 * hi;
  const _Float16* brow3 = Wh + (size_t)(n0 + 48 + ln16) * D_MODEL + 16 * hi;

  v8f acc[4] = {v8f{}, v8f{}, v8f{}, v8f{}};

  // prologue: fetch k0 = 0 fragments
  v16h cb0 = load_h16(brow0);
  v16h cb1 = load_h16(brow1);
  v16h cb2 = load_h16(brow2);
  v16h cb3 = load_h16(brow3);

  for (int k0 = 0; k0 < D_MODEL; k0 += 32) {
    // prefetch next iteration's B fragments (final-iter overread stays
    // inside the workspace carve and is never consumed)
    const int kn = k0 + 32;
    v16h nb0 = load_h16(brow0 + kn);
    v16h nb1 = load_h16(brow1 + kn);
    v16h nb2 = load_h16(brow2 + kn);
    v16h nb3 = load_h16(brow3 + kn);
    // A fragment from LDS (short latency)
    v16h a = load_h16(arow + k0 + 16 * hi);
    acc[0] = __builtin_amdgcn_wmma_f32_16x16x32_f16(false, a, false, cb0, (short)0, acc[0], false, false);
    acc[1] = __builtin_amdgcn_wmma_f32_16x16x32_f16(false, a, false, cb1, (short)0, acc[1], false, false);
    acc[2] = __builtin_amdgcn_wmma_f32_16x16x32_f16(false, a, false, cb2, (short)0, acc[2], false, false);
    acc[3] = __builtin_amdgcn_wmma_f32_16x16x32_f16(false, a, false, cb3, (short)0, acc[3], false, false);
    cb0 = nb0; cb1 = nb1; cb2 = nb2; cb3 = nb3;
  }

  // C/D layout: VGPR r -> row m0 + r + 8*hi, col n0 + t*16 + lane%16
#pragma unroll
  for (int t = 0; t < 4; ++t) {
    const int c = n0 + t * 16 + ln16;
    const int h = c / DK, d = c % DK;
    const float bv = bias[c];
#pragma unroll
    for (int r = 0; r < 8; ++r) {
      const int g = m0 + r + 8 * hi;            // global row in B*L
      const int b = g / L_SEQ, l = g % L_SEQ;
      const size_t idx = (((size_t)(b * NHEAD + h)) * L_SEQ + l) * DK + d;
      const float val = acc[t][r] + bv;
      outf[idx] = val;
      if (outh) outh[idx] = (_Float16)val;
    }
  }
}

// ---- 2) Sampled QK -> M[b,h,l] = max_s(q.k_s) - sum_s(q.k_s)/L ----
__global__ void __launch_bounds__(256)
msample_kernel(const float* __restrict__ q, const float* __restrict__ k,
               const long long* __restrict__ idxs, float* __restrict__ Mbuf) {
  const int tid = blockIdx.x * 256 + threadIdx.x;
  if (tid >= NBH * L_SEQ) return;
  const int bh = tid / L_SEQ, l = tid % L_SEQ;
  const float* qrow = q + ((size_t)bh * L_SEQ + l) * DK;
  float qreg[DK];
#pragma unroll
  for (int d = 0; d < DK; ++d) qreg[d] = qrow[d];
  float mx = -3.0e38f, sm = 0.0f;
  for (int s = 0; s < SAMPK; ++s) {
    const int ki = (int)idxs[(size_t)l * SAMPK + s];
    const float* krow = k + ((size_t)bh * L_SEQ + ki) * DK;
    float dot = 0.0f;
#pragma unroll
    for (int d = 0; d < DK; ++d) dot += qreg[d] * krow[d];
    mx = fmaxf(mx, dot);
    sm += dot;
  }
  Mbuf[tid] = mx - sm / (float)L_SEQ;
}

// ---- 3) Top-40 per (b,h) via iterative argmax in LDS ----
__global__ void __launch_bounds__(256)
topk_kernel(const float* __restrict__ Mbuf, int* __restrict__ Mtop) {
  __shared__ float vals[L_SEQ];
  __shared__ float rmax[256];
  __shared__ int   ridx[256];
  const int bh = blockIdx.x, tid = threadIdx.x;
  for (int i = tid; i < L_SEQ; i += 256) vals[i] = Mbuf[(size_t)bh * L_SEQ + i];
  __syncthreads();
  for (int it = 0; it < N_U; ++it) {
    float best = -3.0e38f; int bi = 0x7fffffff;
    for (int i = tid; i < L_SEQ; i += 256) {
      float v = vals[i];
      if (v > best || (v == best && i < bi)) { best = v; bi = i; }
    }
    rmax[tid] = best; ridx[tid] = bi;
    __syncthreads();
    for (int s = 128; s > 0; s >>= 1) {
      if (tid < s) {
        if (rmax[tid + s] > rmax[tid] ||
            (rmax[tid + s] == rmax[tid] && ridx[tid + s] < ridx[tid])) {
          rmax[tid] = rmax[tid + s]; ridx[tid] = ridx[tid + s];
        }
      }
      __syncthreads();
    }
    if (tid == 0) { Mtop[bh * N_U + it] = ridx[0]; vals[ridx[0]] = -3.0e38f; }
    __syncthreads();
  }
}

// ---- 4) scores = scale * Q_reduce @ k^T   (WMMA f16, 40x2048 per head) ----
__global__ void __launch_bounds__(32)
scores_wmma_kernel(const _Float16* __restrict__ qh, const _Float16* __restrict__ kh,
                   const int* __restrict__ Mtop, float* __restrict__ scores) {
  const int lane = threadIdx.x & 31;
  const int hi = lane >> 4, ln16 = lane & 15;
  const int bh = blockIdx.z;
  const int u0 = blockIdx.y * 16;
  const int n0 = blockIdx.x * 16;
  int uidx = u0 + ln16; if (uidx >= N_U) uidx = N_U - 1;   // clamp pad rows
  const int qrow = Mtop[bh * N_U + uidx];
  const _Float16* arow = qh + ((size_t)bh * L_SEQ + qrow) * DK;
  const _Float16* brow = kh + ((size_t)bh * L_SEQ + (n0 + ln16)) * DK;
  v8f acc = {};
#pragma unroll
  for (int k0 = 0; k0 < DK; k0 += 32) {
    v16h a = load_h16(arow + k0 + 16 * hi);
    v16h b = load_h16(brow + k0 + 16 * hi);
    acc = __builtin_amdgcn_wmma_f32_16x16x32_f16(false, a, false, b, (short)0, acc, false, false);
  }
  const float scale = 0.125f;  // 1/sqrt(dk)
#pragma unroll
  for (int r = 0; r < 8; ++r) {
    const int u = u0 + r + 8 * hi;
    if (u < N_U)
      scores[((size_t)bh * N_U + u) * L_SEQ + n0 + ln16] = acc[r] * scale;
  }
}

// ---- 5) row softmax over L for each of the 640 score rows ----
__global__ void __launch_bounds__(256)
softmax_kernel(float* __restrict__ scores) {
  __shared__ float red[256];
  float* s = scores + (size_t)blockIdx.x * L_SEQ;
  const int tid = threadIdx.x;
  float mx = -3.0e38f;
  for (int i = tid; i < L_SEQ; i += 256) mx = fmaxf(mx, s[i]);
  red[tid] = mx; __syncthreads();
  for (int st = 128; st > 0; st >>= 1) {
    if (tid < st) red[tid] = fmaxf(red[tid], red[tid + st]);
    __syncthreads();
  }
  mx = red[0]; __syncthreads();
  float sm = 0.0f;
  for (int i = tid; i < L_SEQ; i += 256) { float e = __expf(s[i] - mx); s[i] = e; sm += e; }
  red[tid] = sm; __syncthreads();
  for (int st = 128; st > 0; st >>= 1) {
    if (tid < st) red[tid] += red[tid + st];
    __syncthreads();
  }
  const float inv = 1.0f / red[0];
  for (int i = tid; i < L_SEQ; i += 256) s[i] *= inv;
}

// ---- 6) upd = attn @ v ----
__global__ void __launch_bounds__(256)
av_kernel(const float* __restrict__ attn, const float* __restrict__ v,
          float* __restrict__ upd) {
  const int tid = blockIdx.x * 256 + threadIdx.x;
  if (tid >= NBH * N_U * DK) return;
  const int d = tid % DK, rest = tid / DK;
  const int u = rest % N_U, bh = rest / N_U;
  const float* arow  = attn + ((size_t)bh * N_U + u) * L_SEQ;
  const float* vbase = v + (size_t)bh * L_SEQ * DK + d;
  float acc = 0.0f;
#pragma unroll 4
  for (int l = 0; l < L_SEQ; ++l) acc += arow[l] * vbase[(size_t)l * DK];
  upd[tid] = acc;
}

// ---- 7) context = cumsum(v, axis=L) ----
__global__ void __launch_bounds__(64)
cumsum_kernel(const float* __restrict__ v, float* __restrict__ ctx) {
  const int bh = blockIdx.x, d = threadIdx.x;
  const float* vb = v + (size_t)bh * L_SEQ * DK + d;
  float* cb = ctx + (size_t)bh * L_SEQ * DK + d;
  float acc = 0.0f;
  for (int l = 0; l < L_SEQ; ++l) { acc += vb[(size_t)l * DK]; cb[(size_t)l * DK] = acc; }
}

// ---- 8) context[b,h,Mtop] = upd ----
__global__ void __launch_bounds__(256)
scatter_ctx_kernel(const float* __restrict__ upd, const int* __restrict__ Mtop,
                   float* __restrict__ ctx) {
  const int tid = blockIdx.x * 256 + threadIdx.x;
  if (tid >= NBH * N_U * DK) return;
  const int d = tid % DK, rest = tid / DK;
  const int u = rest % N_U, bh = rest / N_U;
  const int l = Mtop[bh * N_U + u];
  ctx[((size_t)bh * L_SEQ + l) * DK + d] = upd[tid];
}

// ---- 9) attns = 1/L everywhere (256 MB stream: vectorized NT stores) ----
__global__ void __launch_bounds__(256)
fill_attns_kernel(f4* __restrict__ p, size_t n4) {
  size_t i = (size_t)blockIdx.x * 256 + threadIdx.x;
  const size_t stride = (size_t)gridDim.x * 256;
  const float fv = 1.0f / (float)L_SEQ;
  f4 val = {fv, fv, fv, fv};
  for (; i < n4; i += stride) __builtin_nontemporal_store(val, p + i);
}

// ---- 10) attns[b,h,Mtop] = attn rows ----
__global__ void __launch_bounds__(256)
scatter_attn_kernel(const float* __restrict__ attn, const int* __restrict__ Mtop,
                    float* __restrict__ attns) {
  const int tid = blockIdx.x * 256 + threadIdx.x;
  if (tid >= NBH * N_U * L_SEQ) return;
  const int l = tid % L_SEQ, rest = tid / L_SEQ;
  const int u = rest % N_U, bh = rest / N_U;
  const int row = Mtop[bh * N_U + u];
  attns[((size_t)bh * L_SEQ + row) * L_SEQ + l] =
      attn[((size_t)bh * N_U + u) * L_SEQ + l];
}

extern "C" void kernel_launch(void* const* d_in, const int* in_sizes, int n_in,
                              void* d_out, int out_size, void* d_ws, size_t ws_size,
                              hipStream_t stream) {
  const float* x  = (const float*)d_in[0];
  const float* Wq = (const float*)d_in[1];
  const float* bq = (const float*)d_in[2];
  const float* Wk = (const float*)d_in[3];
  const float* bk = (const float*)d_in[4];
  const float* Wv = (const float*)d_in[5];
  const float* bv = (const float*)d_in[6];
  const long long* idxs = (const long long*)d_in[7];

  float* out_ctx   = (float*)d_out;            // (B,H,L,dk) flat == reshape(B,L,-1)
  float* out_attns = out_ctx + CTX_ELEMS;      // (B,H,L,L)

  // workspace carve (~45 MB)
  float* qb      = (float*)d_ws;                 // 2,097,152 f32
  float* kb      = qb + CTX_ELEMS;
  float* vb2     = kb + CTX_ELEMS;
  float* Mbuf    = vb2 + CTX_ELEMS;              // 32,768
  float* scoresb = Mbuf + NBH * L_SEQ;           // 1,310,720
  float* updb    = scoresb + NBH * N_U * L_SEQ;  // 40,960
  int*   Mtop    = (int*)(updb + NBH * N_U * DK);// 640 ints (pad to 1024)
  _Float16* xh  = (_Float16*)(Mtop + 1024);      // 2,097,152 halfs
  _Float16* Wqh = xh + (size_t)NROWS * D_MODEL;  // 262,144 halfs each
  _Float16* Wkh = Wqh + D_MODEL * D_MODEL;
  _Float16* Wvh = Wkh + D_MODEL * D_MODEL;
  _Float16* qh  = Wvh + D_MODEL * D_MODEL;       // 2,097,152 halfs
  _Float16* kh  = qh + CTX_ELEMS;

  // 0) f32 -> f16 staging of x and weights
  cvt_f16_kernel<<<(NROWS * D_MODEL + 255) / 256, 256, 0, stream>>>(x, xh, NROWS * D_MODEL);
  cvt_f16_kernel<<<(D_MODEL * D_MODEL + 255) / 256, 256, 0, stream>>>(Wq, Wqh, D_MODEL * D_MODEL);
  cvt_f16_kernel<<<(D_MODEL * D_MODEL + 255) / 256, 256, 0, stream>>>(Wk, Wkh, D_MODEL * D_MODEL);
  cvt_f16_kernel<<<(D_MODEL * D_MODEL + 255) / 256, 256, 0, stream>>>(Wv, Wvh, D_MODEL * D_MODEL);

  // 1) projections (4 waves/block, async-LDS staged A tile, pipelined B)
  const dim3 pgrid(D_MODEL / 256, NROWS / 16);   // (2, 256)
  proj_wmma_kernel<<<pgrid, 128, 0, stream>>>(xh, Wqh, bq, qb, qh);
  proj_wmma_kernel<<<pgrid, 128, 0, stream>>>(xh, Wkh, bk, kb, kh);
  proj_wmma_kernel<<<pgrid, 128, 0, stream>>>(xh, Wvh, bv, vb2, (_Float16*)nullptr);

  msample_kernel<<<(NBH * L_SEQ) / 256, 256, 0, stream>>>(qb, kb, idxs, Mbuf);
  topk_kernel<<<NBH, 256, 0, stream>>>(Mbuf, Mtop);

  scores_wmma_kernel<<<dim3(L_SEQ / 16, 3, NBH), 32, 0, stream>>>(qh, kh, Mtop, scoresb);
  softmax_kernel<<<NBH * N_U, 256, 0, stream>>>(scoresb);
  av_kernel<<<(NBH * N_U * DK + 255) / 256, 256, 0, stream>>>(scoresb, vb2, updb);

  cumsum_kernel<<<NBH, 64, 0, stream>>>(vb2, out_ctx);
  scatter_ctx_kernel<<<(NBH * N_U * DK + 255) / 256, 256, 0, stream>>>(updb, Mtop, out_ctx);

  fill_attns_kernel<<<4096, 256, 0, stream>>>((f4*)out_attns, ATTNS_ELEMS / 4);
  scatter_attn_kernel<<<(NBH * N_U * L_SEQ + 255) / 256, 256, 0, stream>>>(scoresb, Mtop, out_attns);
}
